// Seq2SeqPGGenerator_69801808495262
// MI455X (gfx1250) — compile-verified, matrix operand order checked by reading, not run on hardware
//
#include <hip/hip_runtime.h>
#include <hip/hip_bf16.h>

// ---------------------------------------------------------------------------
// Problem constants (from reference)
// ---------------------------------------------------------------------------
#define B_    32
#define T_    64
#define L_    20
#define V_    20000
#define P_    50
#define C_    100
#define WD_   100
#define NF_   100
#define H_    512
#define G4H_  2048          // 4*H
#define PADTOK 1

// WMMA tiling
#define KT_ENC 26           // ceil((300+512)/32)
#define KT_DEC 36           // ceil((612+512)/32)
#define KT_OUT 16           // 512/32
#define NT_GATE 128         // 2048/16
#define NT_OUT  1250        // 20000/16
#define NT_PER_WAVE 10      // 1250 / 125 waves

typedef __attribute__((ext_vector_type(16))) __bf16 v16bf;
typedef __attribute__((ext_vector_type(8)))  float  v8f;

// ---------------------------------------------------------------------------
// Small device helpers
// ---------------------------------------------------------------------------
__device__ __forceinline__ unsigned short f2bf(float f) {
    unsigned int u = __float_as_uint(f);
    unsigned int r = u + 0x7FFFu + ((u >> 16) & 1u);   // round-to-nearest-even
    return (unsigned short)(r >> 16);
}

__device__ __forceinline__ float sigf(float x) {
    return 1.0f / (1.0f + __expf(-x));
}

// order-preserving (float,col) packing for 64-bit atomicMax argmax
__device__ __forceinline__ unsigned long long packfc(float v, int col) {
    unsigned int u = __float_as_uint(v);
    u = (u & 0x80000000u) ? ~u : (u | 0x80000000u);
    return ((unsigned long long)u << 32) | (unsigned int)(~(unsigned int)col);
}
__device__ __forceinline__ float unpackf(unsigned long long p) {
    unsigned int u = (unsigned int)(p >> 32);
    u = (u & 0x80000000u) ? (u & 0x7FFFFFFFu) : ~u;
    return __uint_as_float(u);
}
__device__ __forceinline__ int unpackc(unsigned long long p) {
    return (int)(~(unsigned int)p);
}

// deterministic hash-based Gumbel noise
__device__ __forceinline__ float gumbelf(int t, int r, int c) {
    unsigned int x = (unsigned int)(t + 1) * 0x9E3779B1u
                   ^ (unsigned int)(r + 7) * 0x85EBCA77u
                   ^ (unsigned int)(c + 13) * 0xC2B2AE3Du;
    x ^= x >> 16; x *= 0x7FEB352Du; x ^= x >> 15; x *= 0x846CA68Bu; x ^= x >> 16;
    float u = (float)(x >> 8) * (1.0f / 16777216.0f) + 1e-7f;
    return -__logf(-__logf(u));
}

// ---------------------------------------------------------------------------
// Weight packer: fp32 row-major [K,N] (optionally two stacked matrices W1;W2,
// zero pad to KT*32) -> bf16 WMMA B-fragment layout:
//   frag (nt,kt): 32 lanes x 16 bf16, lane -> col = nt*16 + (lane&15),
//   element e -> k = kt*32 + (lane&16 ? 8:0) + (e<8 ? e : e+8)
// ---------------------------------------------------------------------------
__global__ void pack_w_kernel(const float* __restrict__ W1, int K1,
                              const float* __restrict__ W2, int K2,
                              int N, int KT, unsigned short* __restrict__ dst) {
    int tile = blockIdx.x;            // tile = nt*KT + kt
    int nt = tile / KT, kt = tile % KT;
    for (int i = threadIdx.x; i < 512; i += blockDim.x) {
        int lane = i >> 4, e = i & 15;
        int col = nt * 16 + (lane & 15);
        int k = kt * 32 + ((lane & 16) ? 8 : 0) + (e < 8 ? e : e + 8);
        float v = 0.0f;
        if (k < K1)                 v = W1[(long)k * N + col];
        else if ((k - K1) < K2)     v = W2[(long)(k - K1) * N + col];
        dst[(long)tile * 512 + i] = f2bf(v);
    }
}

// ---------------------------------------------------------------------------
// Embedding + char conv/maxpool/tanh kernel: one block per (b,t).
// Writes inp[b*T+t][0:100]=word, [100:200]=char feature, [200:300]=pos.
// ---------------------------------------------------------------------------
__global__ void embed_kernel(const int* __restrict__ input_word,
                             const int* __restrict__ input_char,
                             const int* __restrict__ input_pos,
                             const float* __restrict__ word_table,
                             const float* __restrict__ pos_table,
                             const float* __restrict__ char_table,
                             const float* __restrict__ conv_w,
                             const float* __restrict__ conv_b,
                             float* __restrict__ inp) {
    __shared__ float chbuf[L_ * C_];     // [l][c], 8 KB
    int bt = blockIdx.x;
    for (int i = threadIdx.x; i < L_ * C_; i += blockDim.x) {
        int l = i / C_, c = i % C_;
        int ci = input_char[bt * L_ + l];
        chbuf[i] = char_table[(long)ci * C_ + c];
    }
    __syncthreads();
    int n = threadIdx.x;
    if (n < NF_) {
        float mx = -3.4e38f;
        for (int p = 0; p < L_ + 2; ++p) {           // 22 output positions
            float a = conv_b[n];
            for (int k = 0; k < 3; ++k) {
                int pos = p + k - 2;
                if (pos >= 0 && pos < L_) {
                    const float* wr = conv_w + (long)n * 300 + k;  // [n][c][k]
                    const float* xr = chbuf + pos * C_;
                    for (int c = 0; c < C_; ++c) a += wr[c * 3] * xr[c];
                }
            }
            mx = fmaxf(mx, a);
        }
        inp[(long)bt * 300 + 100 + n] = tanhf(mx);
    }
    if (n < 100) {
        int wI = input_word[bt], pI = input_pos[bt];
        inp[(long)bt * 300 + n]       = word_table[(long)wI * WD_ + n];
        inp[(long)bt * 300 + 200 + n] = pos_table[(long)pI * WD_ + n];
    }
}

// ---------------------------------------------------------------------------
// Zero-init of recurrent state / accumulators / argmax slots.
// ---------------------------------------------------------------------------
__global__ void init_kernel(float* h, float* c, float* last, float* acc,
                            unsigned long long* slots) {
    int i = blockIdx.x * blockDim.x + threadIdx.x;
    if (i < B_ * H_) { h[i] = 0.0f; c[i] = 0.0f; }
    if (i < B_ * WD_) last[i] = 0.0f;
    if (i < B_)       acc[i] = 0.0f;
    if (i < 2 * B_)   slots[i] = 0ull;
}

// ---------------------------------------------------------------------------
// Fused gate GEMM: gates[32 x 2048] = concat(src0,src1,src2) @ packedW + bias
// A (32 x KT*32) packed to bf16 WMMA fragments in LDS, one N-tile per wave,
// 2 bf16 WMMAs per K-tile (two M-tiles share the B fragment). KT is a
// compile-time constant so the K loop unrolls and fragment addressing folds
// into immediate offsets; prefetch is unconditional (speculative, branch-free).
// ---------------------------------------------------------------------------
template<int KT>
__global__ void gate_gemm_kernel(const float* __restrict__ src0, int K0, int s0,
                                 const float* __restrict__ src1, int K1, int s1,
                                 const float* __restrict__ src2, int K2, int s2,
                                 const unsigned short* __restrict__ packW,
                                 const float* __restrict__ bias,
                                 float* __restrict__ gates,
                                 int Ntiles) {
    extern __shared__ unsigned short ldsA[];       // [2][KT][32][16] bf16
    constexpr int TOT = 2 * KT * 512;
    for (int i = threadIdx.x; i < TOT; i += 256) {
        int e = i & 15, lane = (i >> 4) & 31;
        int kt = (i >> 9) % KT, mt = (i >> 9) / KT;
        int row = mt * 16 + (lane & 15);
        int k = kt * 32 + ((lane & 16) ? 8 : 0) + (e < 8 ? e : e + 8);
        float v = 0.0f;
        if (k < K0)                      v = src0[(long)row * s0 + k];
        else if ((k - K0) < K1)          v = src1[(long)row * s1 + (k - K0)];
        else if ((k - K0 - K1) < K2)     v = src2[(long)row * s2 + (k - K0 - K1)];
        ldsA[i] = f2bf(v);
    }
    __syncthreads();

    int lane = threadIdx.x & 31;
    int nt = blockIdx.x * 8 + (threadIdx.x >> 5);
    if (nt >= Ntiles) return;

    const v16bf* bAll = (const v16bf*)packW;
    long bbase = (long)nt * KT * 32 + lane;
    v8f c0 = {}, c1 = {};
#pragma unroll 4
    for (int kt = 0; kt < KT; ++kt) {
        v16bf a0 = *(const v16bf*)&ldsA[(size_t)(kt * 32 + lane) * 16];
        v16bf a1 = *(const v16bf*)&ldsA[(size_t)((KT + kt) * 32 + lane) * 16];
        v16bf b  = bAll[bbase + kt * 32];
        __builtin_prefetch(&bAll[bbase + (kt + 1) * 32], 0, 0);  // speculative
        c0 = __builtin_amdgcn_wmma_f32_16x16x32_bf16(false, a0, false, b,
                                                     (short)0, c0, false, false);
        c1 = __builtin_amdgcn_wmma_f32_16x16x32_bf16(false, a1, false, b,
                                                     (short)0, c1, false, false);
    }

    int col = nt * 16 + (lane & 15);
    float bs = bias[col];
    int N = Ntiles * 16;
    int rbase = (lane >> 4) << 3;                   // 0 or 8
#pragma unroll
    for (int v = 0; v < 8; ++v) {
        gates[(long)(v + rbase)      * N + col] = c0[v] + bs;
        gates[(long)(v + rbase + 16) * N + col] = c1[v] + bs;
    }
}

// ---------------------------------------------------------------------------
// Masked LSTM cell elementwise update (in-place h,c); optionally records h.
// ---------------------------------------------------------------------------
__global__ void lstm_cell_kernel(const float* __restrict__ gates,
                                 const float* __restrict__ mask, int t,
                                 float* __restrict__ h, float* __restrict__ cst,
                                 float* __restrict__ hs_out) {
    int i = blockIdx.x * blockDim.x + threadIdx.x;
    if (i >= B_ * H_) return;
    int b = i >> 9, j = i & (H_ - 1);
    const float* g = gates + (long)b * G4H_;
    float gi = g[j], gf = g[H_ + j], gg = g[2 * H_ + j], go = g[3 * H_ + j];
    float cn = sigf(gf) * cst[i] + sigf(gi) * tanhf(gg);
    float hn = sigf(go) * tanhf(cn);
    float m = mask[b * T_ + t];
    float hv = m * hn + (1.0f - m) * h[i];
    float cv = m * cn + (1.0f - m) * cst[i];
    h[i] = hv; cst[i] = cv;
    if (hs_out) hs_out[i] = hv;
}

// ---------------------------------------------------------------------------
// Output projection logits[32 x 20000] = h @ packedOutW + out_b, fused with
// per-row argmax (crit) and gumbel-perturbed argmax (samp) via u64 atomicMax.
// Each wave owns 10 N-tiles; running maxima in registers, 1 atomic per slot.
// ---------------------------------------------------------------------------
__global__ void logits_kernel(const float* __restrict__ h,
                              const unsigned short* __restrict__ packW,
                              const float* __restrict__ outb,
                              unsigned long long* __restrict__ critSlot,
                              unsigned long long* __restrict__ sampSlot,
                              int t) {
    __shared__ unsigned short ldsA[2 * KT_OUT * 512];   // 32 KB
    for (int i = threadIdx.x; i < 2 * KT_OUT * 512; i += 256) {
        int e = i & 15, lane = (i >> 4) & 31;
        int kt = (i >> 9) % KT_OUT, mt = (i >> 9) / KT_OUT;
        int row = mt * 16 + (lane & 15);
        int k = kt * 32 + ((lane & 16) ? 8 : 0) + (e < 8 ? e : e + 8);
        ldsA[i] = f2bf(h[(long)row * H_ + k]);
    }
    __syncthreads();

    int lane = threadIdx.x & 31;
    int gw = blockIdx.x * 8 + (threadIdx.x >> 5);
    if (gw >= NT_OUT / NT_PER_WAVE) return;            // 125 active waves

    const v16bf* bAll = (const v16bf*)packW;
    float bcv[16], bsv[16];
    int   bci[16], bsi[16];
#pragma unroll
    for (int s = 0; s < 16; ++s) { bcv[s] = -3.4e38f; bsv[s] = -3.4e38f; bci[s] = 0; bsi[s] = 0; }

    int rlo = (lane >> 4) << 3;                        // 0 or 8
    for (int nt0 = 0; nt0 < NT_PER_WAVE; ++nt0) {
        int nt = gw * NT_PER_WAVE + nt0;
        long bbase = (long)nt * KT_OUT * 32 + lane;
        v8f c0 = {}, c1 = {};
#pragma unroll 4
        for (int kt = 0; kt < KT_OUT; ++kt) {
            v16bf a0 = *(const v16bf*)&ldsA[(size_t)(kt * 32 + lane) * 16];
            v16bf a1 = *(const v16bf*)&ldsA[(size_t)((KT_OUT + kt) * 32 + lane) * 16];
            v16bf b = bAll[bbase + kt * 32];
            __builtin_prefetch(&bAll[bbase + (kt + 1) * 32], 0, 0);  // speculative
            c0 = __builtin_amdgcn_wmma_f32_16x16x32_bf16(false, a0, false, b,
                                                         (short)0, c0, false, false);
            c1 = __builtin_amdgcn_wmma_f32_16x16x32_bf16(false, a1, false, b,
                                                         (short)0, c1, false, false);
        }
        int col = nt * 16 + (lane & 15);
        float bs = outb[col];
#pragma unroll
        for (int v = 0; v < 8; ++v) {
            int r0 = v + rlo, r1 = r0 + 16;
            float l0 = c0[v] + bs, l1 = c1[v] + bs;
            if (l0 > bcv[v])     { bcv[v] = l0;     bci[v] = col; }
            if (l1 > bcv[8 + v]) { bcv[8 + v] = l1; bci[8 + v] = col; }
            float g0 = l0 + gumbelf(t, r0, col);
            float g1 = l1 + gumbelf(t, r1, col);
            if (g0 > bsv[v])     { bsv[v] = g0;     bsi[v] = col; }
            if (g1 > bsv[8 + v]) { bsv[8 + v] = g1; bsi[8 + v] = col; }
        }
    }
#pragma unroll
    for (int v = 0; v < 8; ++v) {
        int r0 = v + rlo, r1 = r0 + 16;
        atomicMax(&critSlot[r0], packfc(bcv[v],     bci[v]));
        atomicMax(&critSlot[r1], packfc(bcv[8 + v], bci[8 + v]));
        atomicMax(&sampSlot[r0], packfc(bsv[v],     bsi[v]));
        atomicMax(&sampSlot[r1], packfc(bsv[8 + v], bsi[8 + v]));
    }
}

// ---------------------------------------------------------------------------
// Decoder step finalize: decode argmax slots, accumulate acc, gather next
// "last" embedding, reset slots for the following step.
// ---------------------------------------------------------------------------
__global__ void dec_finalize_kernel(unsigned long long* __restrict__ critSlot,
                                    unsigned long long* __restrict__ sampSlot,
                                    const float* __restrict__ mask, int t,
                                    const float* __restrict__ word_table,
                                    float* __restrict__ last,
                                    float* __restrict__ acc,
                                    int* __restrict__ samples,
                                    int* __restrict__ critics) {
    __shared__ int ssam[B_];
    int tid = threadIdx.x;
    if (tid < B_) {
        unsigned long long pc = critSlot[tid], ps = sampSlot[tid];
        float cv = unpackf(pc);
        int ci = unpackc(pc), si = unpackc(ps);
        samples[t * B_ + tid] = si;
        critics[t * B_ + tid] = ci;
        acc[tid] += mask[tid * T_ + t] * cv;
        ssam[tid] = si;
        critSlot[tid] = 0ull;
        sampSlot[tid] = 0ull;
    }
    __syncthreads();
    for (int i = tid; i < B_ * WD_; i += blockDim.x) {
        int b = i / WD_, j = i % WD_;
        last[i] = word_table[(long)ssam[b] * WD_ + j];
    }
}

// ---------------------------------------------------------------------------
// Final output assembly: masked word/char/pos tuples (as floats) + acc.
// Layout (floats): word_s[2048] char_s[40960] pos[2048]
//                  word_c[2048] char_c[40960] pos[2048] acc[32]
// ---------------------------------------------------------------------------
__global__ void emit_kernel(const int* __restrict__ samples,
                            const int* __restrict__ critics,
                            const float* __restrict__ mask,
                            const int* __restrict__ input_pos,
                            const int* __restrict__ lut,
                            const float* __restrict__ acc,
                            float* __restrict__ out) {
    int i = blockIdx.x * blockDim.x + threadIdx.x;
    if (i < B_ * T_) {
        int b = i / T_, t = i % T_;
        bool mb = mask[i] > 0.0f;
        int s = samples[t * B_ + b], c = critics[t * B_ + b];
        int ws = mb ? s : PADTOK, wc = mb ? c : PADTOK;
        out[i]            = (float)ws;
        out[45056 + i]    = (float)wc;
        out[43008 + i]    = (float)input_pos[i];
        out[88064 + i]    = (float)input_pos[i];
        for (int l = 0; l < L_; ++l) {
            out[2048  + i * L_ + l] = (float)lut[(long)ws * L_ + l];
            out[47104 + i * L_ + l] = (float)lut[(long)wc * L_ + l];
        }
    }
    if (i < B_) out[90112 + i] = acc[i];
}

// ---------------------------------------------------------------------------
// Host-side orchestration (single stream, graph-capturable).
// ---------------------------------------------------------------------------
extern "C" void kernel_launch(void* const* d_in, const int* in_sizes, int n_in,
                              void* d_out, int out_size, void* d_ws, size_t ws_size,
                              hipStream_t stream) {
    const int*   input_word = (const int*)d_in[0];
    const int*   input_char = (const int*)d_in[1];
    const int*   input_pos  = (const int*)d_in[2];
    const float* mask       = (const float*)d_in[3];
    const float* word_table = (const float*)d_in[4];
    const float* pos_table  = (const float*)d_in[5];
    const float* char_table = (const float*)d_in[6];
    const float* conv_w     = (const float*)d_in[7];
    const float* conv_b     = (const float*)d_in[8];
    const float* enc_Wi     = (const float*)d_in[9];
    const float* enc_Wh     = (const float*)d_in[10];
    const float* enc_b      = (const float*)d_in[11];
    const float* dec_Wi     = (const float*)d_in[12];
    const float* dec_Wh     = (const float*)d_in[13];
    const float* dec_b      = (const float*)d_in[14];
    const float* out_W      = (const float*)d_in[15];
    const float* out_b      = (const float*)d_in[16];
    const int*   lut        = (const int*)d_in[17];
    float* out = (float*)d_out;

    // workspace carve-up
    char* ws = (char*)d_ws;
    size_t off = 0;
    auto carve = [&](size_t bytes) -> char* {
        char* p = ws + off;
        off = (off + bytes + 255) & ~(size_t)255;
        return p;
    };
    unsigned short* packEnc = (unsigned short*)carve((size_t)NT_GATE * KT_ENC * 512 * 2);
    unsigned short* packDec = (unsigned short*)carve((size_t)NT_GATE * KT_DEC * 512 * 2);
    unsigned short* packOut = (unsigned short*)carve((size_t)NT_OUT * KT_OUT * 512 * 2);
    float* inp     = (float*)carve((size_t)B_ * T_ * 300 * 4);
    float* encbuf  = (float*)carve((size_t)T_ * B_ * H_ * 4);
    float* h       = (float*)carve((size_t)B_ * H_ * 4);
    float* cst     = (float*)carve((size_t)B_ * H_ * 4);
    float* gates   = (float*)carve((size_t)B_ * G4H_ * 4);
    float* last    = (float*)carve((size_t)B_ * WD_ * 4);
    float* acc     = (float*)carve((size_t)B_ * 4);
    int*   samples = (int*)carve((size_t)T_ * B_ * 4);
    int*   critics = (int*)carve((size_t)T_ * B_ * 4);
    unsigned long long* slots = (unsigned long long*)carve(2 * B_ * 8);
    unsigned long long* critSlot = slots;
    unsigned long long* sampSlot = slots + B_;

    // 1) one-time (per call) weight packing fp32 -> bf16 WMMA fragments
    pack_w_kernel<<<NT_GATE * KT_ENC, 256, 0, stream>>>(enc_Wi, 300, enc_Wh, H_,
                                                        G4H_, KT_ENC, packEnc);
    pack_w_kernel<<<NT_GATE * KT_DEC, 256, 0, stream>>>(dec_Wi, 612, dec_Wh, H_,
                                                        G4H_, KT_DEC, packDec);
    pack_w_kernel<<<NT_OUT * KT_OUT, 256, 0, stream>>>(out_W, H_, nullptr, 0,
                                                       V_, KT_OUT, packOut);

    // 2) embeddings + char CNN
    embed_kernel<<<B_ * T_, 128, 0, stream>>>(input_word, input_char, input_pos,
                                              word_table, pos_table, char_table,
                                              conv_w, conv_b, inp);
    init_kernel<<<(B_ * H_ + 255) / 256, 256, 0, stream>>>(h, cst, last, acc, slots);

    // 3) encoder masked LSTM, 64 sequential steps
    for (int t = 0; t < T_; ++t) {
        gate_gemm_kernel<KT_ENC><<<NT_GATE / 8, 256, 2 * KT_ENC * 512 * 2, stream>>>(
            inp + (long)t * 300, 300, T_ * 300,
            h, H_, H_,
            nullptr, 0, 0,
            packEnc, enc_b, gates, NT_GATE);
        lstm_cell_kernel<<<(B_ * H_ + 255) / 256, 256, 0, stream>>>(
            gates, mask, t, h, cst, encbuf + (long)t * B_ * H_);
    }

    // 4) decoder: gate GEMM -> cell -> logits GEMM + fused argmax -> finalize
    for (int t = 0; t < T_; ++t) {
        gate_gemm_kernel<KT_DEC><<<NT_GATE / 8, 256, 2 * KT_DEC * 512 * 2, stream>>>(
            encbuf + (long)t * B_ * H_, H_, H_,
            last, WD_, WD_,
            h, H_, H_,
            packDec, dec_b, gates, NT_GATE);
        lstm_cell_kernel<<<(B_ * H_ + 255) / 256, 256, 0, stream>>>(
            gates, mask, t, h, cst, nullptr);
        logits_kernel<<<16, 256, 0, stream>>>(h, packOut, out_b,
                                              critSlot, sampSlot, t);
        dec_finalize_kernel<<<1, 128, 0, stream>>>(critSlot, sampSlot, mask, t,
                                                   word_table, last, acc,
                                                   samples, critics);
    }

    // 5) assemble outputs
    emit_kernel<<<(B_ * T_ + 255) / 256, 256, 0, stream>>>(samples, critics, mask,
                                                           input_pos, lut, acc, out);
}
